// BoundaryLoss_34961033790042
// MI455X (gfx1250) — matrix operand order repaired and checked by reading, not compiled, not run on hardware
//
#include <hip/hip_runtime.h>
#include <hip/hip_bf16.h>

// ---------------------------------------------------------------------------
// BoundaryLoss for MI455X (gfx1250, wave32).
//   B=8, C=20, H=W=384.  Bandwidth-bound: pred = 94.4 MB read once (~4 us at
//   23.3 TB/s).  EDT done with O(H) column sweeps + O(W^2)-per-row min-plus
//   from LDS (906M VALU ops total -- negligible).  CE stage uses
//   v_wmma_f32_16x16x32_f16: one-hot-gather matmul + ones matmul per
//   16-pixel tile (channel reduction as a real matrix op).
// ---------------------------------------------------------------------------

#define Bn   8
#define Cn   20
#define Hn   384
#define Wn   384
#define HWn  (Hn * Wn)            // 147456
#define Nv   (Bn * HWn)           // 1179648 pixels
#define BIGD 1.0e6f
#define THETA0 3.0f
#define THETA  5.0f

typedef __attribute__((ext_vector_type(16))) _Float16 v16h;
typedef __attribute__((ext_vector_type(8)))  float    v8f;

// ---------------------------------------------------------------------------
// Kernel 1: vertical nearest-zero distance for both masks via 2 sweeps.
//   d1[g][b][y][x] = min_{y'} |y-y'| where t[b][y'][x] == g   (g in {0,1}),
//   clamped to BIG.  One thread per (g,b,x) column; lanes stride x so each
//   y-step is a fully coalesced 32-lane read.
// ---------------------------------------------------------------------------
__global__ __launch_bounds__(256) void edt_cols_kernel(const int* __restrict__ target,
                                                       float* __restrict__ d1) {
  int tid = blockIdx.x * blockDim.x + threadIdx.x;
  if (tid >= 2 * Bn * Wn) return;
  int g = tid / (Bn * Wn);        // mask g: zero-set is (t == g)
  int r = tid - g * (Bn * Wn);
  int b = r / Wn, x = r - b * Wn;
  const int* t = target + (size_t)b * HWn + x;
  float* out = d1 + (size_t)g * Nv + (size_t)b * HWn + x;
  float d = BIGD;
  for (int y = 0; y < Hn; ++y) {
    d = (t[y * Wn] == g) ? 0.0f : fminf(d + 1.0f, BIGD);
    out[y * Wn] = d;
  }
  d = BIGD;
  for (int y = Hn - 1; y >= 0; --y) {
    d = (t[y * Wn] == g) ? 0.0f : fminf(d + 1.0f, BIGD);
    out[y * Wn] = fminf(out[y * Wn], d);
  }
}

// ---------------------------------------------------------------------------
// Kernel 2: exact squared EDT along rows + boundary weight.
//   D[y,x] = min_{x'} d1[y,x']^2 + (x-x')^2 ; w = 1 + 5*exp(-(sqrt(Da)+sqrt(Db))/3)
//   One block per (b,y); both masks' rows staged in LDS; inner loop is a
//   broadcast LDS read (conflict-free) + fma + min.
// ---------------------------------------------------------------------------
__global__ __launch_bounds__(Wn) void edt_rows_kernel(const float* __restrict__ d1,
                                                      float* __restrict__ wmap) {
  __shared__ float sA[Wn];
  __shared__ float sB[Wn];
  const int row = blockIdx.x;     // b*H + y
  const int x = threadIdx.x;
  float va = d1[(size_t)row * Wn + x];
  float vb = d1[(size_t)Nv + (size_t)row * Wn + x];
  sA[x] = va * va;
  sB[x] = vb * vb;
  __syncthreads();
  float ma = 3.4e38f, mb = 3.4e38f;
  #pragma unroll 4
  for (int xp = 0; xp < Wn; ++xp) {
    float dx = (float)(x - xp);
    float dx2 = dx * dx;
    ma = fminf(ma, sA[xp] + dx2);
    mb = fminf(mb, sB[xp] + dx2);
  }
  float dist = sqrtf(ma) + sqrtf(mb);
  wmap[(size_t)row * Wn + x] = 1.0f + THETA * __expf(-dist * (1.0f / THETA0));
}

// ---------------------------------------------------------------------------
// Kernel 3: streaming CE with WMMA channel reduction.
//   Each wave owns a tile of 16 consecutive pixels (tiles never cross batch:
//   HW % 16 == 0).  Logit tile (16 px x 20 ch) staged in LDS, A-matrix is
//   E[m][k] = exp(l[m][k] - max[m]) in f16 (channels >= 20 are exact zeros).
//   WMMA #1: B = per-pixel one-hot(target)  -> diag(D1) = exp(l_t - max)
//   WMMA #2: B = ones                       -> D2[m][*] = sum_k exp(...)
//   ce[m] = log(D2[m]) - log(diag(D1)[m]);  block writes one partial sum.
//
//   A layout (16-bit A 16x32, ISA 7.12.2): lane L holds M = L&15;
//     lanes 0-15: half h -> K = (h<8 ? h : h+8)        (K in 0-7,16-23)
//     lanes 16-31: half h -> K = 8 + (h<8 ? h : h+8)   (K in 8-15,24-31)
//   B layout (32x16): lane holds column N = L&15; half h -> K = (L<16?0:8) + (h<8 ? h : h+8)
//   D layout (f32 16x16): lane holds column N = L&15; vgpr j -> M = j + (L<16?0:8)
// ---------------------------------------------------------------------------
#define WPB  8      // waves per block
#define TPX  16     // pixels per wave
#define CE_BLOCKS (Nv / (WPB * TPX))   // 9216

__global__ __launch_bounds__(256) void ce_wmma_kernel(const float* __restrict__ pred,
                                                      const int* __restrict__ target,
                                                      const float* __restrict__ wmap,
                                                      float* __restrict__ partials) {
  __shared__ float lt[WPB][Cn * TPX];    // logit tiles, channel-major
  __shared__ float dd[WPB][TPX * TPX];   // D-matrix bounce buffer
  __shared__ float mx[WPB][TPX];
  __shared__ int   tg[WPB][TPX];
  __shared__ float wsum[WPB];

  const int wave = threadIdx.x >> 5;
  const int lane = threadIdx.x & 31;
  const long p0 = ((long)blockIdx.x * WPB + wave) * TPX;
  const int b   = (int)(p0 / HWn);
  const int hw0 = (int)(p0 - (long)b * HWn);
  const float* base = pred + (size_t)b * Cn * HWn + hw0;

  // Stage 16x20 logit tile (coalesced: 16 consecutive floats per channel).
  for (int idx = lane; idx < Cn * TPX; idx += 32) {
    int c = idx >> 4, i = idx & 15;
    lt[wave][idx] = base[(size_t)c * HWn + i];
  }
  if (lane < TPX) tg[wave][lane] = target[p0 + lane];
  __syncthreads();

  if (lane < TPX) {
    float m = -3.4e38f;
    for (int c = 0; c < Cn; ++c) m = fmaxf(m, lt[wave][c * 16 + lane]);
    mx[wave][lane] = m;
  }
  __syncthreads();

  // Pack A = exp(l - rowmax), f16, zeros beyond channel 19.
  const int m = lane & 15;
  const int kA = (lane < 16) ? 0 : 8;
  const float mym = mx[wave][m];
  v16h a;
  #pragma unroll
  for (int h = 0; h < 16; ++h) {
    int k = kA + (h < 8 ? h : h + 8);
    float e = (k < Cn) ? __expf(lt[wave][k * 16 + m] - mym) : 0.0f;
    a[h] = (_Float16)e;
  }
  // Pack B1 (one-hot of target[n]) and B2 (ones on valid channels).
  const int n = lane & 15;
  const int kB = (lane < 16) ? 0 : 8;
  const int tn = tg[wave][n];
  v16h b1, b2;
  #pragma unroll
  for (int h = 0; h < 16; ++h) {
    int k = kB + (h < 8 ? h : h + 8);
    b1[h] = (_Float16)((k == tn) ? 1.0f : 0.0f);
    b2[h] = (_Float16)((k < Cn) ? 1.0f : 0.0f);
  }
  v8f c0 = {};
  // 8 args: (neg_a, A, neg_b, B, c_mod, C, reuse_a, reuse_b)
  v8f dg = __builtin_amdgcn_wmma_f32_16x16x32_f16(false, a, false, b1, (short)0, c0, false, false);
  v8f ds = __builtin_amdgcn_wmma_f32_16x16x32_f16(false, a, false, b2, (short)0, c0, false, false);

  // Bounce D1 through LDS, extract diagonal (gathered exp values).
  const int Mb = (lane < 16) ? 0 : 8;
  #pragma unroll
  for (int j = 0; j < 8; ++j) dd[wave][(Mb + j) * 16 + n] = dg[j];
  __syncthreads();
  float etgt = (lane < TPX) ? dd[wave][lane * 16 + lane] : 0.0f;
  __syncthreads();
  #pragma unroll
  for (int j = 0; j < 8; ++j) dd[wave][(Mb + j) * 16 + n] = ds[j];
  __syncthreads();

  float acc = 0.0f;
  if (lane < TPX) {
    float se = dd[wave][lane * 16 + 0];            // every column of D2 equal
    float ce = __logf(se) - __logf(fmaxf(etgt, 1e-30f));
    acc = ce * wmap[p0 + lane];
  }
  // Wave reduce (lanes >= 16 carry 0; lane 0 result is clean).
  #pragma unroll
  for (int off = 16; off >= 1; off >>= 1) acc += __shfl_down(acc, off, 32);
  if (lane == 0) wsum[wave] = acc;
  __syncthreads();
  if (threadIdx.x == 0) {
    float s = 0.0f;
    #pragma unroll
    for (int i = 0; i < WPB; ++i) s += wsum[i];
    partials[blockIdx.x] = s;
  }
}

// ---------------------------------------------------------------------------
// Kernel 4: deterministic final reduction (fixed per-thread order + fixed tree)
// ---------------------------------------------------------------------------
__global__ __launch_bounds__(256) void final_reduce_kernel(const float* __restrict__ partials,
                                                           int np, float* __restrict__ out) {
  __shared__ float s[256];
  float acc = 0.0f;
  for (int i = threadIdx.x; i < np; i += 256) acc += partials[i];
  s[threadIdx.x] = acc;
  __syncthreads();
  for (int off = 128; off >= 1; off >>= 1) {
    if ((int)threadIdx.x < off) s[threadIdx.x] += s[threadIdx.x + off];
    __syncthreads();
  }
  if (threadIdx.x == 0) out[0] = s[0] / (float)Nv;
}

// ---------------------------------------------------------------------------
extern "C" void kernel_launch(void* const* d_in, const int* in_sizes, int n_in,
                              void* d_out, int out_size, void* d_ws, size_t ws_size,
                              hipStream_t stream) {
  (void)in_sizes; (void)n_in; (void)out_size; (void)ws_size;
  const float* pred   = (const float*)d_in[0];   // (B,C,H,W) f32
  const int*   target = (const int*)d_in[1];     // (B,H,W)   i32
  float* out = (float*)d_out;                    // scalar mean

  // Workspace layout (floats): [d1: 2*Nv][wmap: Nv][partials: CE_BLOCKS]
  float* wsf      = (float*)d_ws;
  float* d1       = wsf;
  float* wmap     = wsf + (size_t)2 * Nv;
  float* partials = wsf + (size_t)3 * Nv;

  // 1) column sweeps: 2*B*W = 6144 threads
  edt_cols_kernel<<<(2 * Bn * Wn + 255) / 256, 256, 0, stream>>>(target, d1);
  // 2) row EDT + weights: one block per (b,y)
  edt_rows_kernel<<<Bn * Hn, Wn, 0, stream>>>(d1, wmap);
  // 3) streaming CE with WMMA channel reduction
  ce_wmma_kernel<<<CE_BLOCKS, 256, 0, stream>>>(pred, target, wmap, partials);
  // 4) deterministic scalar reduction
  final_reduce_kernel<<<1, 256, 0, stream>>>(partials, CE_BLOCKS, out);
}
